// EdgeGate_6786048328249
// MI455X (gfx1250) — matrix operand order repaired and checked by reading, not compiled, not run on hardware
//
#include <hip/hip_runtime.h>
#include <stdint.h>

typedef __attribute__((ext_vector_type(16))) __bf16 v16bf;
typedef __attribute__((ext_vector_type(2)))  __bf16 v2bf;
typedef __attribute__((ext_vector_type(8)))  float  v8f;

#define N_EDGES 600000
#define D_MODEL 128
#define HIDDEN  128
#define TILE_E  64
#define ROWELEM 264   /* 256 bf16 + 8 pad -> 528B row stride (16B multiple, staggers banks) */
#define NTILES  ((N_EDGES + TILE_E - 1) / TILE_E)

union V16BFu { v16bf v; uint4 q[2]; };
union V2BFu  { v2bf  v; unsigned int u; };

__device__ __forceinline__ unsigned int pack2bf(float a, float b) {
  // Plain casts -> backend picks hardware cvt (v_cvt_pk_bf16_f32 if available)
  V2BFu r;
  r.v = (v2bf){(__bf16)a, (__bf16)b};
  return r.u;
}

__global__ __launch_bounds__(256, 1) void edge_gate_wmma(
    const float* __restrict__ H, const int* __restrict__ ei,
    const float* __restrict__ S, const float* __restrict__ F,
    const float* __restrict__ W1, const float* __restrict__ b1,
    const float* __restrict__ W2, const float* __restrict__ b2,
    float* __restrict__ out)
{
  __shared__ __align__(16) unsigned short smA[TILE_E * ROWELEM];
  __shared__ float2 smSF[TILE_E];
  __shared__ float  smSum[TILE_E];

  const int tid  = threadIdx.x;
  const int lane = tid & 31;
  const int wave = tid >> 5;      // 0..7 -> owns HIDDEN columns [wave*16, wave*16+16)
  const int half = lane >> 4;     // 0,1
  const int l15  = lane & 15;

  // ---- per-wave B fragments of W1 (bf16), kept in VGPRs for the whole kernel ----
  // B 32x16 bf16 layout: lane%16 = N, element j -> K = 8*half + j (j<8), 16 + 8*half + (j-8) (j>=8).
  const int ncol = wave * 16;
  v16bf bfrag[8];
  #pragma unroll
  for (int kt = 0; kt < 8; ++kt) {
    const float* wrow = W1 + (kt * 32 + half * 8) * HIDDEN + ncol + l15;
    v16bf t;
    #pragma unroll
    for (int j = 0; j < 16; ++j) t[j] = (__bf16)wrow[(j < 8 ? j : j + 8) * HIDDEN];
    bfrag[kt] = t;
  }
  // per-lane fp32 tail constants for column n = ncol + l15
  const int   n     = ncol + l15;
  const float w1s   = W1[256 * HIDDEN + n];   // S_e row of W1
  const float w1f   = W1[257 * HIDDEN + n];   // F_e row of W1
  const float b1n   = b1[n];
  const float w2n   = W2[n];
  const float bias2 = b2[0];

  for (int tile = blockIdx.x; tile < NTILES; tile += (int)gridDim.x) {
    const int base = tile * TILE_E;

    // ---- gather endpoint features, build bf16 A tile [64 x 256] in LDS ----
    {
      const int e_loc = tid >> 2;       // 0..63
      const int q     = tid & 3;        // quarter of the 128-dim row
      const int edge  = base + e_loc;
      const int u = ei[edge];
      const int v = ei[N_EDGES + edge];
      const float4* Hu = (const float4*)(H + (long)u * D_MODEL);
      const float4* Hv = (const float4*)(H + (long)v * D_MODEL);
      unsigned short* row = smA + e_loc * ROWELEM;
      #pragma unroll
      for (int i = 0; i < 8; ++i) {
        const int d4 = q * 8 + i;                 // float4 index within 128-dim row
        float4 a  = Hu[d4];
        float4 bb = Hv[d4];
        float dx = fabsf(a.x - bb.x), dy = fabsf(a.y - bb.y);
        float dz = fabsf(a.z - bb.z), dw = fabsf(a.w - bb.w);
        float px = a.x * bb.x, py = a.y * bb.y;
        float pz = a.z * bb.z, pw = a.w * bb.w;
        uint2 dpk; dpk.x = pack2bf(dx, dy); dpk.y = pack2bf(dz, dw);
        uint2 ppk; ppk.x = pack2bf(px, py); ppk.y = pack2bf(pz, pw);
        *(uint2*)(row + d4 * 4)       = dpk;      // K = 0..127   : |Hu-Hv|
        *(uint2*)(row + 128 + d4 * 4) = ppk;      // K = 128..255 : Hu*Hv
      }
      if (q == 0) smSF[e_loc] = make_float2(S[edge], F[edge]);
      if (tid < TILE_E) smSum[tid] = 0.0f;
    }
    __syncthreads();

    // ---- layer 1: [64 x 256]bf16 x [256 x 16]bf16 -> fp32, per wave ----
    v8f acc[4];
    #pragma unroll
    for (int mt = 0; mt < 4; ++mt) acc[mt] = (v8f)(0.0f);

    const int cbase = half * 8;   // A layout: lane half selects K offset 0 or 8
    #pragma unroll
    for (int kt = 0; kt < 8; ++kt) {
      #pragma unroll
      for (int mt = 0; mt < 4; ++mt) {
        const unsigned short* p = smA + (mt * 16 + l15) * ROWELEM + kt * 32 + cbase;
        V16BFu af;
        af.q[0] = *(const uint4*)(p);        // K = c .. c+7
        af.q[1] = *(const uint4*)(p + 16);   // K = c+16 .. c+23
        acc[mt] = __builtin_amdgcn_wmma_f32_16x16x32_bf16(
            false, af.v, false, bfrag[kt], (short)0, acc[mt], false, false);
      }
    }

    // ---- fp32 tail (S,F,b1) + SiLU + layer 2 (dot with W2) + cross-wave reduce ----
    #pragma unroll
    for (int mt = 0; mt < 4; ++mt) {
      #pragma unroll
      for (int r = 0; r < 8; ++r) {
        const int e_loc = mt * 16 + r + half * 8;   // C/D layout: vgpr r -> M=r (+8 hi half)
        float2 sf  = smSF[e_loc];
        float pre  = acc[mt][r] + sf.x * w1s + sf.y * w1f + b1n;
        float hact = pre * (1.0f / (1.0f + __expf(-pre)));   // SiLU
        float part = hact * w2n;
        part += __shfl_xor(part, 1);
        part += __shfl_xor(part, 2);
        part += __shfl_xor(part, 4);
        part += __shfl_xor(part, 8);
        if (l15 == 0) atomicAdd(&smSum[e_loc], part);        // ds_add_f32 across waves
      }
    }
    __syncthreads();

    if (tid < TILE_E) out[base + tid] = smSum[tid] + bias2;
    __syncthreads();
  }
}

extern "C" void kernel_launch(void* const* d_in, const int* in_sizes, int n_in,
                              void* d_out, int out_size, void* d_ws, size_t ws_size,
                              hipStream_t stream) {
  const float* H  = (const float*)d_in[0];
  const int*   ei = (const int*)d_in[1];
  const float* S  = (const float*)d_in[2];
  const float* F  = (const float*)d_in[3];
  const float* W1 = (const float*)d_in[4];
  const float* b1 = (const float*)d_in[5];
  const float* W2 = (const float*)d_in[6];
  const float* b2 = (const float*)d_in[7];
  float* out = (float*)d_out;

  dim3 grid(2048), block(256);
  edge_gate_wmma<<<grid, block, 0, stream>>>(H, ei, S, F, W1, b1, W2, b2, out);
}
